// MoEEncoderDecoderGPT_15126874817031
// MI455X (gfx1250) — matrix operand order repaired and checked by reading, not compiled, not run on hardware
//
#include <hip/hip_runtime.h>
#include <hip/hip_bf16.h>
#include <hip/hip_fp16.h>

typedef _Float16 v16h __attribute__((ext_vector_type(16)));
typedef _Float16 v8h  __attribute__((ext_vector_type(8)));
typedef float    v8f  __attribute__((ext_vector_type(8)));

#define DN 1024   // n_embd
#define HD 2048   // hidden
#define AD 128    // adapt dim
#define NE 8      // experts
#define NB 4      // batch
#define SS 4096   // seq
#define NT (NB*SS)

__device__ __forceinline__ v16h cat16(v8h lo, v8h hi) {
    return __builtin_shufflevector(lo, hi, 0,1,2,3,4,5,6,7,8,9,10,11,12,13,14,15);
}
__device__ __forceinline__ float dsilu(float x) { return x / (1.0f + __expf(-x)); }

#define LDSW 40  // padded half-stride: 80B rows keep 16B alignment, stagger banks

// ---------------------------------------------------------------------------
// f16-operand GEMM: A:(M,K) f16, B:(N,K) f16, 256x128x32 block, 8 waves 64x64.
// Double-buffered LDS: next tile's global loads issued before current tile's
// WMMAs, ds_store after compute, one barrier per k-step.
// MODE: 0 C=alpha*acc, 1 C+=alpha*acc, 3 f16-only, 4 C = X + alpha*acc.
// WH: also write f16 mirror Ch.
// ---------------------------------------------------------------------------
template<int MODE, bool WH>
__global__ __launch_bounds__(256) void gemm_wmma_h(
    const _Float16* __restrict__ A, const _Float16* __restrict__ B,
    float* __restrict__ C, _Float16* __restrict__ Ch,
    const float* __restrict__ X, int M, int N, int K, float alpha)
{
    __shared__ _Float16 As[2][256 * LDSW];  // 2 x 20KB
    __shared__ _Float16 Bs[2][128 * LDSW];  // 2 x 10KB
    const int t   = threadIdx.x;
    const int m0  = blockIdx.y * 256;
    const int n0  = blockIdx.x * 128;
    const int w   = t >> 5, L = t & 31;
    const int hi  = L >> 4, l16 = L & 15;
    const int wm  = (w & 3) * 64;
    const int wn  = (w >> 2) * 64;

    v8f acc[4][4];
#pragma unroll
    for (int i = 0; i < 4; i++)
#pragma unroll
        for (int j = 0; j < 4; j++) acc[i][j] = (v8f){};

    const _Float16* arow = A + (size_t)(m0 + t) * K;
    const _Float16* brow = B + (size_t)(n0 + (t & 127)) * K;

    v8h ar[4], br[4];
    {   // tile 0 -> regs -> buf 0
        const v8h* s = (const v8h*)arow;
#pragma unroll
        for (int q = 0; q < 4; q++) ar[q] = s[q];
        v8h* d = (v8h*)&As[0][t * LDSW];
#pragma unroll
        for (int q = 0; q < 4; q++) d[q] = ar[q];
        if (t < 128) {
            const v8h* sb = (const v8h*)brow;
#pragma unroll
            for (int q = 0; q < 4; q++) br[q] = sb[q];
            v8h* db = (v8h*)&Bs[0][t * LDSW];
#pragma unroll
            for (int q = 0; q < 4; q++) db[q] = br[q];
        }
    }
    __syncthreads();

    const int nk = K >> 5;
    for (int kt = 0; kt < nk; kt++) {
        const int cur = kt & 1, nxt = cur ^ 1;
        const bool more = (kt + 1) < nk;
        if (more) {   // issue next tile's loads before computing (overlap)
            const v8h* s = (const v8h*)(arow + (kt + 1) * 32);
#pragma unroll
            for (int q = 0; q < 4; q++) ar[q] = s[q];
            if (t < 128) {
                const v8h* sb = (const v8h*)(brow + (kt + 1) * 32);
#pragma unroll
                for (int q = 0; q < 4; q++) br[q] = sb[q];
            }
            if (kt + 2 < nk) {   // deeper prefetch toward WGP caches
                __builtin_prefetch(arow + (kt + 2) * 32, 0, 1);
                if (t < 128) __builtin_prefetch(brow + (kt + 2) * 32, 0, 1);
            }
        }
        v16h af[4];
#pragma unroll
        for (int i = 0; i < 4; i++) {
            const int mrow = wm + i * 16 + l16;
            af[i] = cat16(*(const v8h*)&As[cur][mrow * LDSW + 8 * hi],
                          *(const v8h*)&As[cur][mrow * LDSW + 16 + 8 * hi]);
        }
#pragma unroll
        for (int j = 0; j < 4; j++) {
            const int nrow = wn + j * 16 + l16;
            v16h bf = cat16(*(const v8h*)&Bs[cur][nrow * LDSW + 8 * hi],
                            *(const v8h*)&Bs[cur][nrow * LDSW + 16 + 8 * hi]);
#pragma unroll
            for (int i = 0; i < 4; i++)
                acc[i][j] = __builtin_amdgcn_wmma_f32_16x16x32_f16(
                    false, af[i], false, bf, (short)0, acc[i][j], false, false);
        }
        if (more) {
            v8h* d = (v8h*)&As[nxt][t * LDSW];
#pragma unroll
            for (int q = 0; q < 4; q++) d[q] = ar[q];
            if (t < 128) {
                v8h* db = (v8h*)&Bs[nxt][t * LDSW];
#pragma unroll
                for (int q = 0; q < 4; q++) db[q] = br[q];
            }
        }
        __syncthreads();
    }
#pragma unroll
    for (int i = 0; i < 4; i++)
#pragma unroll
        for (int j = 0; j < 4; j++)
#pragma unroll
            for (int jj = 0; jj < 8; jj++) {
                size_t row = (size_t)(m0 + wm + i * 16 + jj + 8 * hi);
                size_t col = (size_t)(n0 + wn + j * 16 + l16);
                size_t idx = row * (size_t)N + col;
                float v;
                if constexpr (MODE == 0)      { v = alpha * acc[i][j][jj]; C[idx] = v; }
                else if constexpr (MODE == 1) { v = C[idx] + alpha * acc[i][j][jj]; C[idx] = v; }
                else if constexpr (MODE == 4) { v = X[idx] + alpha * acc[i][j][jj]; C[idx] = v; }
                else                          { v = alpha * acc[i][j][jj]; }
                if constexpr (WH) Ch[idx] = (_Float16)v;
            }
}

// ---------------------------------------------------------------------------
// Fused dual-B GEMM: hidden = silu(x@Wg^T) * (x@Wu^T), f32 + f16 outputs.
// Block 128x128x32, 8 waves of 32x64; removes a full hidden write+read pass.
// ---------------------------------------------------------------------------
__global__ __launch_bounds__(256) void gemm_dual_silu(
    const _Float16* __restrict__ A, const _Float16* __restrict__ Bu,
    const _Float16* __restrict__ Bg, float* __restrict__ C,
    _Float16* __restrict__ Ch, int M, int N, int K)
{
    __shared__ _Float16 As[128 * LDSW];
    __shared__ _Float16 Us[128 * LDSW];
    __shared__ _Float16 Gs[128 * LDSW];
    const int t   = threadIdx.x;
    const int m0  = blockIdx.y * 128;
    const int n0  = blockIdx.x * 128;
    const int w   = t >> 5, L = t & 31;
    const int hi  = L >> 4, l16 = L & 15;
    const int wm  = (w & 3) * 32;
    const int wn  = (w >> 2) * 64;
    const int r   = t & 127;

    v8f aU[2][4], aG[2][4];
#pragma unroll
    for (int i = 0; i < 2; i++)
#pragma unroll
        for (int j = 0; j < 4; j++) { aU[i][j] = (v8f){}; aG[i][j] = (v8f){}; }

    const _Float16* arow = A  + (size_t)(m0 + r) * K;
    const _Float16* urow = Bu + (size_t)(n0 + r) * K;
    const _Float16* grow = Bg + (size_t)(n0 + r) * K;

    for (int k0 = 0; k0 < K; k0 += 32) {
        if (t < 128) {
            const v8h* s = (const v8h*)(arow + k0);
            v8h* d = (v8h*)&As[r * LDSW];
#pragma unroll
            for (int q = 0; q < 4; q++) d[q] = s[q];
            const v8h* sg = (const v8h*)(grow + k0);
            v8h* dg = (v8h*)&Gs[r * LDSW];
#pragma unroll
            for (int q = 0; q < 4; q++) dg[q] = sg[q];
        } else {
            const v8h* su = (const v8h*)(urow + k0);
            v8h* du = (v8h*)&Us[r * LDSW];
#pragma unroll
            for (int q = 0; q < 4; q++) du[q] = su[q];
        }
        if (k0 + 32 < K) {
            __builtin_prefetch(arow + k0 + 32, 0, 1);
            __builtin_prefetch((t < 128 ? grow : urow) + k0 + 32, 0, 1);
        }
        __syncthreads();
        v16h af[2];
#pragma unroll
        for (int i = 0; i < 2; i++) {
            const int mrow = wm + i * 16 + l16;
            af[i] = cat16(*(const v8h*)&As[mrow * LDSW + 8 * hi],
                          *(const v8h*)&As[mrow * LDSW + 16 + 8 * hi]);
        }
#pragma unroll
        for (int j = 0; j < 4; j++) {
            const int nrow = wn + j * 16 + l16;
            v16h bu = cat16(*(const v8h*)&Us[nrow * LDSW + 8 * hi],
                            *(const v8h*)&Us[nrow * LDSW + 16 + 8 * hi]);
            v16h bg = cat16(*(const v8h*)&Gs[nrow * LDSW + 8 * hi],
                            *(const v8h*)&Gs[nrow * LDSW + 16 + 8 * hi]);
#pragma unroll
            for (int i = 0; i < 2; i++) {
                aU[i][j] = __builtin_amdgcn_wmma_f32_16x16x32_f16(
                    false, af[i], false, bu, (short)0, aU[i][j], false, false);
                aG[i][j] = __builtin_amdgcn_wmma_f32_16x16x32_f16(
                    false, af[i], false, bg, (short)0, aG[i][j], false, false);
            }
        }
        __syncthreads();
    }
#pragma unroll
    for (int i = 0; i < 2; i++)
#pragma unroll
        for (int j = 0; j < 4; j++)
#pragma unroll
            for (int jj = 0; jj < 8; jj++) {
                size_t row = (size_t)(m0 + wm + i * 16 + jj + 8 * hi);
                size_t col = (size_t)(n0 + wn + j * 16 + l16);
                size_t idx = row * (size_t)N + col;
                float v = dsilu(aG[i][j][jj]) * aU[i][j][jj];
                C[idx]  = v;
                Ch[idx] = (_Float16)v;
            }
}

// ---------------------------------------------------------------------------
// Fused adapter "attention": adapt[s,:] = sum_t silu(clip(<Q[s],K[t]>)) * V[t,:]
// Fragments preloaded into register arrays so loads clause/overlap the WMMAs.
// ---------------------------------------------------------------------------
__global__ __launch_bounds__(128) void flash_adapt(
    const _Float16* __restrict__ Qh, const _Float16* __restrict__ Kh,
    const _Float16* __restrict__ VT, _Float16* __restrict__ Out)
{
    __shared__ _Float16 Ws[4][16 * 32];
    const int t  = threadIdx.x;
    const int w  = t >> 5, L = t & 31, hi = L >> 4, l16 = L & 15;
    const int b  = blockIdx.y;
    const int s0 = blockIdx.x * 64 + w * 16;

    v16h qf[4];
    {
        const _Float16* qrow = Qh + (size_t)(b * SS + s0 + l16) * AD;
#pragma unroll
        for (int kc = 0; kc < 4; kc++)
            qf[kc] = cat16(*(const v8h*)(qrow + kc * 32 + 8 * hi),
                           *(const v8h*)(qrow + kc * 32 + 16 + 8 * hi));
    }
    v8f out[8];
#pragma unroll
    for (int c = 0; c < 8; c++) out[c] = (v8f){};

    for (int t0 = 0; t0 < SS; t0 += 32) {
        if (t0 + 32 < SS) {
            __builtin_prefetch(Kh + (size_t)(b * SS + t0 + 32 + l16) * AD + 16 * hi, 0, 1);
            __builtin_prefetch(VT + ((size_t)(b * AD + hi * 64 + l16 * 4)) * SS + t0 + 32, 0, 1);
        }
        // preload all 8 K-fragments first (loads clause together)
        v16h kf[2][4];
#pragma unroll
        for (int sub = 0; sub < 2; sub++) {
            const _Float16* krow = Kh + (size_t)(b * SS + t0 + sub * 16 + l16) * AD;
#pragma unroll
            for (int kc = 0; kc < 4; kc++)
                kf[sub][kc] = cat16(*(const v8h*)(krow + kc * 32 + 8 * hi),
                                    *(const v8h*)(krow + kc * 32 + 16 + 8 * hi));
        }
        v8f sc[2];
#pragma unroll
        for (int sub = 0; sub < 2; sub++) {
            v8f a = (v8f){};
#pragma unroll
            for (int kc = 0; kc < 4; kc++)
                a = __builtin_amdgcn_wmma_f32_16x16x32_f16(
                    false, qf[kc], false, kf[sub][kc], (short)0, a, false, false);
#pragma unroll
            for (int jj = 0; jj < 8; jj++) {
                float x = fminf(5.0f, fmaxf(-5.0f, a[jj]));
                a[jj] = dsilu(x);
            }
            sc[sub] = a;
        }
        // bounce score tile (D layout) -> A-fragment layout via per-wave LDS
#pragma unroll
        for (int sub = 0; sub < 2; sub++)
#pragma unroll
            for (int jj = 0; jj < 8; jj++)
                Ws[w][(jj + 8 * hi) * 32 + sub * 16 + l16] = (_Float16)sc[sub][jj];
        v16h sf = cat16(*(const v8h*)&Ws[w][l16 * 32 + 8 * hi],
                        *(const v8h*)&Ws[w][l16 * 32 + 16 + 8 * hi]);
        // V fragments in chunks of 4 (loads overlap preceding WMMAs)
#pragma unroll
        for (int cc = 0; cc < 8; cc += 4) {
            v16h vf[4];
#pragma unroll
            for (int q = 0; q < 4; q++) {
                const _Float16* vrow =
                    VT + ((size_t)(b * AD + (cc + q) * 16 + l16)) * SS + t0;
                vf[q] = cat16(*(const v8h*)(vrow + 8 * hi),
                              *(const v8h*)(vrow + 16 + 8 * hi));
            }
#pragma unroll
            for (int q = 0; q < 4; q++)
                out[cc + q] = __builtin_amdgcn_wmma_f32_16x16x32_f16(
                    false, sf, false, vf[q], (short)0, out[cc + q], false, false);
        }
    }
#pragma unroll
    for (int c = 0; c < 8; c++)
#pragma unroll
        for (int jj = 0; jj < 8; jj++)
            Out[(size_t)(b * SS + s0 + jj + 8 * hi) * AD + c * 16 + l16] =
                (_Float16)out[c][jj];
}

// ---------------------------------------------------------------------------
// Router: logits=LN(x@W^T) -> softmax(T) -> top-2 dispatch; atomic load/z sums
// ---------------------------------------------------------------------------
__global__ __launch_bounds__(256) void router_kernel(
    const float* __restrict__ x,  const float* __restrict__ rw,
    const float* __restrict__ g,  const float* __restrict__ bb,
    const float* __restrict__ temp, float* __restrict__ dispatch,
    float* __restrict__ loadsum, float* __restrict__ zsum)
{
    const int t = threadIdx.x;
    const int w = t >> 5, L = t & 31;
    const int n = blockIdx.x * 8 + w;
    float acc[NE];
#pragma unroll
    for (int e = 0; e < NE; e++) acc[e] = 0.0f;
    const float* xr = x + (size_t)n * DN;
    for (int d = L; d < DN; d += 32) {
        float xv = xr[d];
#pragma unroll
        for (int e = 0; e < NE; e++) acc[e] += xv * rw[e * DN + d];
    }
#pragma unroll
    for (int e = 0; e < NE; e++)
#pragma unroll
        for (int off = 16; off > 0; off >>= 1)
            acc[e] += __shfl_xor(acc[e], off, 32);
    float m = 0.0f;
#pragma unroll
    for (int e = 0; e < NE; e++) m += acc[e];
    m *= (1.0f / NE);
    float var = 0.0f;
#pragma unroll
    for (int e = 0; e < NE; e++) { float d = acc[e] - m; var += d * d; }
    var *= (1.0f / NE);
    float inv = rsqrtf(var + 1e-5f);
    float ln[NE], zp = 0.0f;
#pragma unroll
    for (int e = 0; e < NE; e++) {
        ln[e] = (acc[e] - m) * inv * g[e] + bb[e];
        zp += ln[e] * ln[e];
    }
    float ti = 1.0f / (temp[0] + 1e-6f);
    float mx = ln[0] * ti;
#pragma unroll
    for (int e = 1; e < NE; e++) mx = fmaxf(mx, ln[e] * ti);
    float p[NE], sum = 0.0f;
#pragma unroll
    for (int e = 0; e < NE; e++) { p[e] = __expf(ln[e] * ti - mx); sum += p[e]; }
    float rs = 1.0f / sum;
#pragma unroll
    for (int e = 0; e < NE; e++) p[e] *= rs;
    int i1 = 0;
#pragma unroll
    for (int e = 1; e < NE; e++) if (p[e] > p[i1]) i1 = e;
    int i2 = (i1 == 0) ? 1 : 0;
#pragma unroll
    for (int e = 0; e < NE; e++) if (e != i1 && p[e] > p[i2]) i2 = e;
    if (L == 0) {
#pragma unroll
        for (int e = 0; e < NE; e++)
            dispatch[(size_t)n * NE + e] = (e == i1) ? p[i1] : ((e == i2) ? p[i2] : 0.0f);
        int bi = n / SS;
        atomicAdd(&loadsum[bi * NE + i1], p[i1]);
        atomicAdd(&loadsum[bi * NE + i2], p[i2]);
        atomicAdd(zsum, zp);
    }
}

// ---------------------------------------------------------------------------
// LayerNorm over width 128; optional f16 out and per-batch transposed f16 out
// ---------------------------------------------------------------------------
__global__ __launch_bounds__(128) void ln128(
    const float* __restrict__ in, const float* __restrict__ g,
    const float* __restrict__ bb, _Float16* out_h, _Float16* out_T)
{
    __shared__ float red[128];
    const int row = blockIdx.x, c = threadIdx.x;
    float v = in[(size_t)row * AD + c];
    red[c] = v; __syncthreads();
    for (int s = 64; s > 0; s >>= 1) { if (c < s) red[c] += red[c + s]; __syncthreads(); }
    float mean = red[0] * (1.0f / AD); __syncthreads();
    float d = v - mean;
    red[c] = d * d; __syncthreads();
    for (int s = 64; s > 0; s >>= 1) { if (c < s) red[c] += red[c + s]; __syncthreads(); }
    float var = red[0] * (1.0f / AD);
    float r = d * rsqrtf(var + 1e-5f) * g[c] + bb[c];
    if (out_h) out_h[(size_t)row * AD + c] = (_Float16)r;
    if (out_T) { int bi = row / SS, s = row % SS; out_T[((size_t)bi * AD + c) * SS + s] = (_Float16)r; }
}

// comb[n,:] += dispatch[n,e] * LN_e(in[n,:]); on last expert also emit f16
__global__ __launch_bounds__(128) void ln_expert_combine(
    const float* __restrict__ in, const float* __restrict__ g,
    const float* __restrict__ bb, const float* __restrict__ dispatch,
    float* __restrict__ comb, _Float16* comb_h, int e)
{
    __shared__ float red[128];
    const int row = blockIdx.x, c = threadIdx.x;
    float v = in[(size_t)row * AD + c];
    red[c] = v; __syncthreads();
    for (int s = 64; s > 0; s >>= 1) { if (c < s) red[c] += red[c + s]; __syncthreads(); }
    float mean = red[0] * (1.0f / AD); __syncthreads();
    float d = v - mean;
    red[c] = d * d; __syncthreads();
    for (int s = 64; s > 0; s >>= 1) { if (c < s) red[c] += red[c + s]; __syncthreads(); }
    float var = red[0] * (1.0f / AD);
    float r = d * rsqrtf(var + 1e-5f) * g[c] + bb[c];
    float wgt = dispatch[(size_t)row * NE + e];
    float nv = comb[(size_t)row * AD + c] + wgt * r;
    comb[(size_t)row * AD + c] = nv;
    if (comb_h) comb_h[(size_t)row * AD + c] = (_Float16)nv;
}

__global__ void zero_f32(float* __restrict__ p, size_t count) {
    for (size_t i = blockIdx.x * (size_t)blockDim.x + threadIdx.x; i < count;
         i += (size_t)gridDim.x * blockDim.x) p[i] = 0.0f;
}
__global__ void cvt_f32_f16(const float* __restrict__ in, _Float16* __restrict__ out, size_t count) {
    for (size_t i = blockIdx.x * (size_t)blockDim.x + threadIdx.x; i < count;
         i += (size_t)gridDim.x * blockDim.x) out[i] = (_Float16)in[i];
}

__global__ void loss_kernel(const float* __restrict__ loadsum,
                            const float* __restrict__ zsum, float* __restrict__ out)
{
    const int t = threadIdx.x;           // 32 threads, one per (b,e)
    float v = loadsum[t] * (1.0f / SS);
    float s = v;
#pragma unroll
    for (int off = 16; off > 0; off >>= 1) s += __shfl_xor(s, off, 32);
    float mean = s * (1.0f / 32.0f);
    float d = v - mean;
    float ss = d * d;
#pragma unroll
    for (int off = 16; off > 0; off >>= 1) ss += __shfl_xor(ss, off, 32);
    if (t == 0) {
        float stddev = sqrtf(ss / 31.0f);          // ddof=1
        float lb = stddev / mean * 10.0f;
        float z = zsum[0] / (float)(NT * NE);
        out[0] = 0.001f * z + 0.1f * lb;
    }
}

extern "C" void kernel_launch(void* const* d_in, const int* in_sizes, int n_in,
                              void* d_out, int out_size, void* d_ws, size_t ws_size,
                              hipStream_t stream) {
    (void)in_sizes; (void)n_in; (void)out_size; (void)ws_size;
    const float* x             = (const float*)d_in[0];
    const float* shared_up_w   = (const float*)d_in[1];
    const float* shared_gate_w = (const float*)d_in[2];
    const float* shared_down_w = (const float*)d_in[3];
    const float* pre_adapt_w   = (const float*)d_in[4];
    const float* post_adapt_w  = (const float*)d_in[5];
    const float* adapt_norm_g  = (const float*)d_in[6];
    const float* adapt_norm_b  = (const float*)d_in[7];
    const float* adapt_proj_w  = (const float*)d_in[8];
    const float* router_w      = (const float*)d_in[9];
    const float* router_ln_g   = (const float*)d_in[10];
    const float* router_ln_b   = (const float*)d_in[11];
    const float* temperature   = (const float*)d_in[12];
    const float* expert_adapt_w= (const float*)d_in[13];
    const float* expert_ln_g   = (const float*)d_in[14];
    const float* expert_ln_b   = (const float*)d_in[15];
    const float* expert_proj_w = (const float*)d_in[16];
    const float* output_proj_w = (const float*)d_in[17];
    float* y = (float*)d_out;

    size_t off = 0;
    auto carve = [&](size_t bytes) -> void* {
        void* p = (void*)((char*)d_ws + off);
        off += (bytes + 255) & ~(size_t)255;
        return p;
    };
    float*     hidden      = (float*)    carve((size_t)NT * HD * 4);
    _Float16*  hidden_h    = (_Float16*) carve((size_t)NT * HD * 2);  // reused as t1h
    _Float16*  xh          = (_Float16*) carve((size_t)NT * DN * 2);
    float*     tmpA        = (float*)    carve((size_t)NT * AD * 4);
    _Float16*  adapt_in_h  = (_Float16*) carve((size_t)NT * AD * 2);
    _Float16*  adapt_out_h = (_Float16*) carve((size_t)NT * AD * 2);
    _Float16*  adapt_in_T  = (_Float16*) carve((size_t)NT * AD * 2);  // [b][a][s]
    _Float16*  adaptv_h    = (_Float16*) carve((size_t)NT * AD * 2);
    float*     comb        = (float*)    carve((size_t)NT * AD * 4);
    _Float16*  comb_h      = (_Float16*) carve((size_t)NT * AD * 2);
    float*     dispatch    = (float*)    carve((size_t)NT * NE * 4);
    float*     loadsum     = (float*)    carve((NB * NE + 1) * 4);
    float*     zsum        = loadsum + NB * NE;
    _Float16*  up_wh       = (_Float16*) carve((size_t)HD * DN * 2);
    _Float16*  gate_wh     = (_Float16*) carve((size_t)HD * DN * 2);
    _Float16*  down_wh     = (_Float16*) carve((size_t)DN * HD * 2);
    _Float16*  pre_wh      = (_Float16*) carve((size_t)AD * DN * 2);
    _Float16*  post_wh     = (_Float16*) carve((size_t)AD * HD * 2);
    _Float16*  aproj_wh    = (_Float16*) carve((size_t)HD * AD * 2);
    _Float16*  exp_wh      = (_Float16*) carve((size_t)NE * AD * AD * 2);
    _Float16*  eproj_wh    = (_Float16*) carve((size_t)HD * AD * 2);
    _Float16*  oproj_wh    = (_Float16*) carve((size_t)DN * HD * 2);

    // zero accumulators ; f16 conversions (one time per operand)
    zero_f32<<<2048, 256, 0, stream>>>(comb, (size_t)NT * AD);
    zero_f32<<<1, 64, 0, stream>>>(loadsum, NB * NE + 1);
    cvt_f32_f16<<<4096, 256, 0, stream>>>(x, xh, (size_t)NT * DN);
    cvt_f32_f16<<<1024, 256, 0, stream>>>(shared_up_w,   up_wh,   (size_t)HD * DN);
    cvt_f32_f16<<<1024, 256, 0, stream>>>(shared_gate_w, gate_wh, (size_t)HD * DN);
    cvt_f32_f16<<<1024, 256, 0, stream>>>(shared_down_w, down_wh, (size_t)DN * HD);
    cvt_f32_f16<<<128,  256, 0, stream>>>(pre_adapt_w,   pre_wh,  (size_t)AD * DN);
    cvt_f32_f16<<<256,  256, 0, stream>>>(post_adapt_w,  post_wh, (size_t)AD * HD);
    cvt_f32_f16<<<256,  256, 0, stream>>>(adapt_proj_w,  aproj_wh,(size_t)HD * AD);
    cvt_f32_f16<<<128,  256, 0, stream>>>(expert_adapt_w,exp_wh,  (size_t)NE * AD * AD);
    cvt_f32_f16<<<256,  256, 0, stream>>>(expert_proj_w, eproj_wh,(size_t)HD * AD);
    cvt_f32_f16<<<1024, 256, 0, stream>>>(output_proj_w, oproj_wh,(size_t)DN * HD);

    // router -> dispatch, load sums, z-loss sum
    router_kernel<<<NT / 8, 256, 0, stream>>>(x, router_w, router_ln_g, router_ln_b,
                                              temperature, dispatch, loadsum, zsum);

    // shared MLP fused: hidden = silu(x@gate^T)*(x@up^T), f32 + f16
    gemm_dual_silu<<<dim3(HD/128, NT/128), 256, 0, stream>>>(
        xh, up_wh, gate_wh, hidden, hidden_h, NT, HD, DN);

    // adapt_in = LN(x@pre^T) -> f16 + per-batch transposed f16 (flash V^T)
    gemm_wmma_h<0,false><<<dim3(1, NT/256), 256, 0, stream>>>(
        xh, pre_wh, tmpA, nullptr, nullptr, NT, AD, DN, 1.0f);
    ln128<<<NT, 128, 0, stream>>>(tmpA, adapt_norm_g, adapt_norm_b, adapt_in_h, adapt_in_T);

    // adapt_out = LN(hidden@post^T) -> f16
    gemm_wmma_h<0,false><<<dim3(1, NT/256), 256, 0, stream>>>(
        hidden_h, post_wh, tmpA, nullptr, nullptr, NT, AD, HD, 1.0f);
    ln128<<<NT, 128, 0, stream>>>(tmpA, adapt_norm_g, adapt_norm_b, adapt_out_h, nullptr);

    // fused adapter attention: adaptv = silu(clip(Q K^T)) @ V   (f16 out)
    flash_adapt<<<dim3(SS/64, NB), 128, 0, stream>>>(adapt_in_h, adapt_out_h, adapt_in_T, adaptv_h);

    // hidden += 0.1*adaptv@aproj^T (emit updated f16) ; y = x + hidden@down^T
    gemm_wmma_h<1,true><<<dim3(HD/128, NT/256), 256, 0, stream>>>(
        adaptv_h, aproj_wh, hidden, hidden_h, nullptr, NT, HD, AD, 0.1f);
    gemm_wmma_h<4,false><<<dim3(DN/128, NT/256), 256, 0, stream>>>(
        hidden_h, down_wh, y, nullptr, x, NT, DN, HD, 1.0f);

    // experts: comb += dispatch[:,e] * LN_e(adapt_in @ W_e^T); last emits f16
    for (int e = 0; e < NE; e++) {
        gemm_wmma_h<0,false><<<dim3(1, NT/256), 256, 0, stream>>>(
            adapt_in_h, exp_wh + (size_t)e * AD * AD, tmpA, nullptr, nullptr, NT, AD, AD, 1.0f);
        ln_expert_combine<<<NT, 128, 0, stream>>>(
            tmpA, expert_ln_g + e * AD, expert_ln_b + e * AD, dispatch, comb,
            (e == NE - 1) ? comb_h : nullptr, e);
    }

    // eo: t1h = comb@eproj^T (f16 only, reuse hidden_h) ; y += t1h@oproj^T
    gemm_wmma_h<3,true><<<dim3(HD/128, NT/256), 256, 0, stream>>>(
        comb_h, eproj_wh, nullptr, hidden_h, nullptr, NT, HD, AD, 1.0f);
    gemm_wmma_h<1,false><<<dim3(DN/128, NT/256), 256, 0, stream>>>(
        hidden_h, oproj_wh, y, nullptr, nullptr, NT, DN, HD, 1.0f);

    // router loss scalar
    loss_kernel<<<1, 32, 0, stream>>>(loadsum, zsum, y + (size_t)NT * DN);
}
